// FARGANSub_23398981829052
// MI455X (gfx1250) — matrix operand order repaired and checked by reading, not compiled, not run on hardware
//
#include <hip/hip_runtime.h>

// ---------------------------------------------------------------------------
// FARGAN subframe network on gfx1250 (MI455X): bf16 WMMA 16x16x32 pipeline
// with bf16 LDS activation mirrors, TDM input staging, NT streaming I/O,
// and LDS trimmed to 157 KB so 2 workgroups fit per 320 KB WGP.
// ---------------------------------------------------------------------------

typedef __attribute__((ext_vector_type(16))) __bf16 v16bf;
typedef __attribute__((ext_vector_type(8)))  __bf16 v8bf;
typedef __attribute__((ext_vector_type(8)))  float  v8f;
typedef __attribute__((ext_vector_type(4)))  unsigned int v4u;
typedef __attribute__((ext_vector_type(8)))  int    v8i;
typedef __attribute__((ext_vector_type(4)))  int    v4i;

#ifndef __has_builtin
#define __has_builtin(x) 0
#endif
#if __has_builtin(__builtin_amdgcn_tensor_load_to_lds)
#  if __has_include(<hip/amd_detail/amd_gfx1250_TDM.h>)
#    define TDM_LOAD(g0, g1, z2, z3) \
        __builtin_amdgcn_tensor_load_to_lds(g0, g1, z2, z3, (v8i){0,0,0,0,0,0,0,0}, 0)
#  else
#    define TDM_LOAD(g0, g1, z2, z3) \
        __builtin_amdgcn_tensor_load_to_lds(g0, g1, z2, z3, 0)
#  endif
#  define HAVE_TDM 1
#else
#  define HAVE_TDM 0
#endif

constexpr int BATCH = 65536;

// ---------------- packed bf16 weight offsets (elements) --------------------
constexpr size_t PK_FWC    = 0;
constexpr size_t PK_FWCGLU = PK_FWC    + (size_t)192*352;
constexpr size_t PK_W1IH   = PK_FWCGLU + (size_t)192*192;
constexpr size_t PK_W1HH   = PK_W1IH   + (size_t)480*288;
constexpr size_t PK_WG1    = PK_W1HH   + (size_t)480*160;
constexpr size_t PK_W2IH   = PK_WG1    + (size_t)160*160;
constexpr size_t PK_W2HH   = PK_W2IH   + (size_t)384*256;
constexpr size_t PK_WG2    = PK_W2HH   + (size_t)384*128;
constexpr size_t PK_W3IH   = PK_WG2    + (size_t)128*128;
constexpr size_t PK_W3HH   = PK_W3IH   + (size_t)384*224;
constexpr size_t PK_WG3    = PK_W3HH   + (size_t)384*128;
constexpr size_t PK_WSKIP  = PK_WG3    + (size_t)128*128;
constexpr size_t PK_WGSKIP = PK_WSKIP  + (size_t)128*704;
constexpr size_t PK_WSIG   = PK_WGSKIP + (size_t)128*128;

// ---------------- output layout (floats) -----------------------------------
constexpr size_t O_SIG = 0;
constexpr size_t O_EXC = (size_t)BATCH * 40;
constexpr size_t O_PP  = (size_t)BATCH * 296;
constexpr size_t O_G1  = (size_t)BATCH * 552;
constexpr size_t O_G2  = (size_t)BATCH * 712;
constexpr size_t O_G3  = (size_t)BATCH * 840;
constexpr size_t O_S4  = (size_t)BATCH * 968;

// ---------------- LDS budget (float slots) ---------------------------------
// persistent: xcat f32(5632) | xcat bf16(2816) | fwc f32(3072) | fwc bf16(1536)
//             o1 bf16(1280) | o2 bf16(1024) | o3 bf16(1024) | gain/rinv/pg(96)
constexpr int SMF_PERSIST = 5632 + 2816 + 3072 + 1536 + 1280 + 1024 + 1024 + 96;
// arena worst case = GRU1: xb(288bf)+hb(160bf)+gi(480f)+gh(480f)+g(160f)+gb(160bf)
// (lin aliases gi; s_exc(4096f) and fwc/skip scratch also live here)
constexpr int SMF_ARENA   = (16*288)/2 + (16*160)/2 + 16*480 + 16*480 + 16*160 + (16*160)/2;
constexpr int SM_FLOATS   = SMF_PERSIST + SMF_ARENA;     // 39264 -> 157056 B
constexpr size_t SMEM_BYTES = (size_t)SM_FLOATS * sizeof(float);

__device__ __forceinline__ float sigmoidf_(float x) { return 1.f / (1.f + __expf(-x)); }

// ---------------------------------------------------------------------------
// Weight packing: fp32 W[N][K] row-major -> bf16 WMMA B-fragment stream.
// ---------------------------------------------------------------------------
template<int N, int K, int NP, int KP>
__global__ void pack_w(const float* __restrict__ W, __bf16* __restrict__ out) {
    constexpr int TOT = NP * KP;
    constexpr int KC  = KP / 32;
    int e = blockIdx.x * 256 + threadIdx.x;
    if (e >= TOT) return;
    int blk = e >> 9;
    int r   = e & 511;
    int L   = r >> 4, j = r & 15;
    int kc  = blk % KC, nt = blk / KC;
    int n   = nt * 16 + (L & 15);
    int k   = kc * 32 + ((L >> 4) << 3) + (j & 7) + ((j >> 3) << 4);
    float v = (n < N && k < K) ? W[(size_t)n * K + k] : 0.f;
    out[e] = (__bf16)v;
}

// ---------------------------------------------------------------------------
// GEMM: C[16][CW] = A_bf16_lds[16][KP] * Wpacked (N cols; cols>=NVALID masked).
// ---------------------------------------------------------------------------
template<int KP, int N, int NVALID, int CW>
__device__ __forceinline__ void mm_bf(const __bf16* __restrict__ A,
                                      const __bf16* __restrict__ Wp,
                                      float* __restrict__ C, int wave, int lane) {
    constexpr int KC = KP / 32;
    constexpr int NT = N / 16;
    const int m  = lane & 15;
    const int hi = lane >> 4;
    const __bf16* arow = A + m * KP;
    for (int nt = wave; nt < NT; nt += 8) {
        v8f acc = {0.f,0.f,0.f,0.f,0.f,0.f,0.f,0.f};
        const __bf16* wp = Wp + (size_t)nt * KC * 512 + (size_t)lane * 16;
        #pragma unroll
        for (int kc = 0; kc < KC; ++kc) {
            const __bf16* pa = arow + kc * 32 + hi * 8;
            v8bf lo = *(const v8bf*)(pa);
            v8bf hh = *(const v8bf*)(pa + 16);
            v16bf a = __builtin_shufflevector(lo, hh, 0,1,2,3,4,5,6,7,8,9,10,11,12,13,14,15);
            v16bf b = *(const v16bf*)(wp + (size_t)kc * 512);
            acc = __builtin_amdgcn_wmma_f32_16x16x32_bf16(false, a, false, b,
                                                          (short)0, acc, false, false);
        }
        const int n = nt * 16 + m;
        if (n < NVALID) {
            #pragma unroll
            for (int v = 0; v < 8; ++v) C[(hi * 8 + v) * CW + n] = acc[v];
        }
    }
}

// One GRU layer + GLU for a 16-row tile; all A operands are bf16 in LDS.
template<int XW, int PO_N, int H>
__device__ __forceinline__ void gru_layer(float* __restrict__ arena,
                                          const __bf16* __restrict__ bprev,
                                          const float* __restrict__ s_xcat,
                                          const float* __restrict__ s_pg, int pg_idx,
                                          const float* __restrict__ h_glob,
                                          const __bf16* __restrict__ Wih,
                                          const __bf16* __restrict__ Whh,
                                          const __bf16* __restrict__ Wg,
                                          float* __restrict__ out_g,
                                          __bf16* __restrict__ b_o,
                                          int tid, int wave, int lane) {
    constexpr int NG = 3 * H;
    __bf16* xb  = (__bf16*)arena;          // 16*XW bf16
    __bf16* hb  = xb + 16 * XW;            // 16*H  bf16
    float*  gi  = (float*)(hb + 16 * H);   // 16*NG f32
    float*  gh  = gi + 16 * NG;            // 16*NG f32
    float*  g   = gh + 16 * NG;            // 16*H  f32
    __bf16* gb  = (__bf16*)(g + 16 * H);   // 16*H  bf16
    float*  lin = gi;                      // aliases gi (dead after combine)

    for (int e = tid; e < 16 * XW; e += 256) {   // x = [prev | pg*fpitch | prev_exc | 0]
        int r = e / XW, c = e % XW; __bf16 v;
        if      (c < PO_N)      v = bprev[r * PO_N + c];
        else if (c < PO_N + 40) v = (__bf16)(s_pg[r * 4 + pg_idx] * s_xcat[r * 352 + 246 + (c - PO_N)]);
        else if (c < PO_N + 80) v = (__bf16)(s_xcat[r * 352 + 288 + (c - PO_N - 40)]);
        else                    v = (__bf16)0.f;
        xb[e] = v;
    }
    for (int e = tid; e < 16 * H; e += 256) hb[e] = (__bf16)h_glob[e];   // stage h once
    __syncthreads();
    mm_bf<XW, NG, NG, NG>(xb, Wih, gi, wave, lane);
    mm_bf<H,  NG, NG, NG>(hb, Whh, gh, wave, lane);
    __syncthreads();
    for (int e = tid; e < 16 * H; e += 256) {
        int r = e / H, c = e % H;
        const float* gir = gi + r * NG;
        const float* ghr = gh + r * NG;
        float rg = sigmoidf_(gir[c]         + ghr[c]);
        float z  = sigmoidf_(gir[H + c]     + ghr[H + c]);
        float n  = tanhf   (gir[2 * H + c] + rg * ghr[2 * H + c]);
        float gv = (1.f - z) * n + z * h_glob[r * H + c];
        g[e]  = gv;
        gb[e] = (__bf16)gv;
        __builtin_nontemporal_store(gv, out_g + e);
    }
    __syncthreads();
    mm_bf<H, H, H, H>(gb, Wg, lin, wave, lane);
    __syncthreads();
    for (int e = tid; e < 16 * H; e += 256) b_o[e] = (__bf16)(g[e] * sigmoidf_(lin[e]));
    __syncthreads();
}

// ---------------------------------------------------------------------------
// Main kernel: 16 batch rows per 256-thread (8-wave, wave32) workgroup.
// ---------------------------------------------------------------------------
__global__ __launch_bounds__(256) void fargan_main(
    const float* __restrict__ cond, const float* __restrict__ prev_pred,
    const float* __restrict__ exc_mem,
    const float* __restrict__ s1, const float* __restrict__ s2,
    const float* __restrict__ s3, const float* __restrict__ s4,
    const int*   __restrict__ period,
    const float* __restrict__ W_cg, const float* __restrict__ b_cg,
    const float* __restrict__ W_pg, const float* __restrict__ b_pg,
    const __bf16* __restrict__ pk, float* __restrict__ out) {
    extern __shared__ float sm[];
    const int tid = threadIdx.x, lane = tid & 31, wave = tid >> 5;
    const int r0 = blockIdx.x * 16;

    float*  s_xcat = sm;                                // 16 x 352 f32
    __bf16* b_xcat = (__bf16*)(s_xcat + 5632);          // 16 x 352 bf16
    float*  s_fwc  = s_xcat + 5632 + 2816;              // 16 x 192 f32
    __bf16* b_fwc  = (__bf16*)(s_fwc + 3072);           // 16 x 192 bf16
    __bf16* b_o1   = b_fwc + 3072;                      // 16 x 160 bf16
    __bf16* b_o2   = b_o1 + 2560;                       // 16 x 128 bf16
    __bf16* b_o3   = b_o2 + 2048;                       // 16 x 128 bf16
    float*  s_gain = s_fwc + 3072 + 1536 + 1280 + 1024 + 1024;
    float*  s_rinv = s_gain + 16;
    float*  s_pg   = s_rinv + 16;
    float*  arena  = s_pg + 64;
    float*  s_exc  = arena;                             // 16 x 256 f32 (phase-0 only)

    float* out_sig = out + O_SIG + (size_t)r0 * 40;
    float* out_exc = out + O_EXC + (size_t)r0 * 256;
    float* out_pp  = out + O_PP  + (size_t)r0 * 256;
    float* out_g1  = out + O_G1  + (size_t)r0 * 160;
    float* out_g2  = out + O_G2  + (size_t)r0 * 128;
    float* out_g3  = out + O_G3  + (size_t)r0 * 128;
    float* out_s4  = out + O_S4  + (size_t)r0 * 164;

    // ---- Phase 0a: kick off TDM load of exc tile (2 rows per wave) --------
#if HAVE_TDM
    {
        const int wu = __builtin_amdgcn_readfirstlane(wave);
        unsigned long long ga = (unsigned long long)(size_t)(const void*)exc_mem
                              + (unsigned long long)(r0 + wu * 2) * 1024ull;
        unsigned int ldsoff = (unsigned int)((const char*)s_exc - (const char*)sm)
                            + (unsigned int)(wu * 2048);
        v4u g0 = { 1u,                                    // count=1, user descriptor
                   ldsoff,                                // lds_addr (bytes)
                   (unsigned int)(ga & 0xffffffffull),    // global_addr[31:0]
                   (unsigned int)((ga >> 32) & 0x01ffffffull) | (2u << 30) }; // addr | type=2
        // group1: data_size=4B; tensor_dim0=256; tensor_dim1=BATCH; tile=256x2; stride0=256
        v8i g1 = { 0x20000, 0x01000000, 0, 0x01000001, 2, 256, 0, 0 };
        v4i gz = { 0, 0, 0, 0 };
        TDM_LOAD(g0, g1, gz, gz);
    }
#else
    for (int e = tid; e < 16 * 256; e += 256)
        s_exc[e] = exc_mem[(size_t)(r0 + (e >> 8)) * 256 + (e & 255)];
#endif

    // ---- Phase 0b: stage xcat = [s4 | cond | pred | prev] (f32 + bf16) ----
    for (int e = tid; e < 16 * 164; e += 256) {
        int r = e / 164, c = e % 164;
        float v = __builtin_nontemporal_load(s4 + (size_t)(r0 + r) * 164 + c);
        s_xcat[r * 352 + c] = v;
        b_xcat[r * 352 + c] = (__bf16)v;
    }
    for (int e = tid; e < 16 * 80; e += 256) {
        int r = e / 80, c = e % 80;
        float v = __builtin_nontemporal_load(cond + (size_t)(r0 + r) * 80 + c);
        s_xcat[r * 352 + 164 + c] = v;
        b_xcat[r * 352 + 164 + c] = (__bf16)v;
    }
    for (int e = tid; e < 16 * 24; e += 256) {
        int r = e / 24, c = e % 24;
        s_xcat[r * 352 + 328 + c] = 0.f;
        b_xcat[r * 352 + 328 + c] = (__bf16)0.f;
    }
#if HAVE_TDM
    __builtin_amdgcn_s_wait_tensorcnt(0);
#endif
    __syncthreads();

    if (tid < 16) {                                   // gain (N=1 -> VALU)
        float acc = b_cg[0];
        const float* cr = s_xcat + tid * 352 + 164;
        for (int k = 0; k < 80; ++k) acc += cr[k] * W_cg[k];
        float g = 0.2f + 0.8f * sigmoidf_(acc);
        g = fminf(fmaxf(g, 0.001f), 20.f);
        s_gain[tid] = g;
        s_rinv[tid] = 1.f / (1e-5f + g);
    }
    __syncthreads();
    for (int e = tid; e < 16 * 44; e += 256) {        // pred gather (from LDS exc)
        int r = e / 44, j = e % 44;
        int per = period[r0 + r];
        int idx = 256 - per + j - 2;
        if (idx >= 256) idx -= per;
        idx = idx < 0 ? 0 : (idx > 255 ? 255 : idx);
        float v = s_exc[r * 256 + idx] * s_rinv[r];
        s_xcat[r * 352 + 244 + j] = v;
        b_xcat[r * 352 + 244 + j] = (__bf16)v;
    }
    for (int e = tid; e < 16 * 40; e += 256) {        // prev = exc[-40:] * rinv
        int r = e / 40, j = e % 40;
        float v = s_exc[r * 256 + 216 + j] * s_rinv[r];
        s_xcat[r * 352 + 288 + j] = v;
        b_xcat[r * 352 + 288 + j] = (__bf16)v;
    }
    for (int e = tid; e < 16 * 216; e += 256) {       // exc_new head = exc[:,40:]
        int r = e / 216, c = e % 216;
        __builtin_nontemporal_store(s_exc[r * 256 + 40 + c], out_exc + r * 256 + c);
    }
    __syncthreads();                                  // s_exc dead; arena free
    for (int e = tid; e < 16 * 164; e += 256) {       // s4_new = tmp (exact fp32)
        int r = e / 164, c = e % 164;
        __builtin_nontemporal_store(s_xcat[r * 352 + 164 + c], out_s4 + e);
    }
    for (int e = tid; e < 16 * 256; e += 256) {       // pp_new = [prev_pred[:,40:], fpitch]
        int r = e / 256, c = e % 256;
        float v = (c < 216) ? __builtin_nontemporal_load(prev_pred + (size_t)(r0 + r) * 256 + 40 + c)
                            : s_xcat[r * 352 + 246 + (c - 216)];
        __builtin_nontemporal_store(v, out_pp + e);
    }

    // ---- Phase 1: fwc0 = glu(tanh(xcat @ W_fwc.T), W_fwc_glu); pitch_gain -
    float*  t_fwc = arena;                     // 16 x 192 f32
    __bf16* tb    = (__bf16*)(t_fwc + 3072);   // 16 x 192 bf16
    float*  lin2  = (float*)(tb + 3072);       // 16 x 192 f32
    mm_bf<352, 192, 192, 192>(b_xcat, pk + PK_FWC, t_fwc, wave, lane);
    __syncthreads();
    for (int e = tid; e < 16 * 192; e += 256) {
        float t = tanhf(t_fwc[e]);
        t_fwc[e] = t;
        tb[e] = (__bf16)t;
    }
    __syncthreads();
    mm_bf<192, 192, 192, 192>(tb, pk + PK_FWCGLU, lin2, wave, lane);
    __syncthreads();
    for (int e = tid; e < 16 * 192; e += 256) {
        float v = t_fwc[e] * sigmoidf_(lin2[e]);
        s_fwc[e] = v;
        b_fwc[e] = (__bf16)v;
    }
    __syncthreads();
    if (tid < 64) {                                   // pitch_gain (N=4 -> VALU)
        int r = tid >> 2, c = tid & 3;
        float acc = b_pg[c];
        const float* fr = s_fwc + r * 192;
        const float* wr = W_pg + c * 192;
        for (int k = 0; k < 192; ++k) acc += fr[k] * wr[k];
        s_pg[r * 4 + c] = sigmoidf_(acc);
    }
    __syncthreads();

    // ---- Phase 2-4: three GRU+GLU layers ----------------------------------
    gru_layer<288, 192, 160>(arena, b_fwc, s_xcat, s_pg, 0, s1 + (size_t)r0 * 160,
                             pk + PK_W1IH, pk + PK_W1HH, pk + PK_WG1,
                             out_g1, b_o1, tid, wave, lane);
    gru_layer<256, 160, 128>(arena, b_o1, s_xcat, s_pg, 1, s2 + (size_t)r0 * 128,
                             pk + PK_W2IH, pk + PK_W2HH, pk + PK_WG2,
                             out_g2, b_o2, tid, wave, lane);
    gru_layer<224, 128, 128>(arena, b_o2, s_xcat, s_pg, 2, s3 + (size_t)r0 * 128,
                             pk + PK_W3IH, pk + PK_W3HH, pk + PK_WG3,
                             out_g3, b_o3, tid, wave, lane);

    // ---- Phase 5: skip = glu(tanh(skip_in @ W_skip.T), Wg_skip) -----------
    __bf16* skinb = (__bf16*)arena;              // 16 x 704 bf16
    float*  tsk   = (float*)(skinb + 16 * 704);  // 16 x 128 f32
    __bf16* tskb  = (__bf16*)(tsk + 2048);       // 16 x 128 bf16
    float*  lsk   = (float*)(tskb + 2048);       // 16 x 128 f32
    for (int e = tid; e < 16 * 704; e += 256) {
        int r = e / 704, c = e % 704; __bf16 v;
        if      (c < 160) v = b_o1[r * 160 + c];
        else if (c < 288) v = b_o2[r * 128 + (c - 160)];
        else if (c < 416) v = b_o3[r * 128 + (c - 288)];
        else if (c < 608) v = b_fwc[r * 192 + (c - 416)];
        else if (c < 648) v = (__bf16)(s_pg[r * 4 + 3] * s_xcat[r * 352 + 246 + (c - 608)]);
        else if (c < 688) v = (__bf16)(s_xcat[r * 352 + 288 + (c - 648)]);
        else              v = (__bf16)0.f;
        skinb[e] = v;
    }
    __syncthreads();
    mm_bf<704, 128, 128, 128>(skinb, pk + PK_WSKIP, tsk, wave, lane);
    __syncthreads();
    for (int e = tid; e < 16 * 128; e += 256) {
        float t = tanhf(tsk[e]);
        tsk[e] = t;
        tskb[e] = (__bf16)t;
    }
    __syncthreads();
    mm_bf<128, 128, 128, 128>(tskb, pk + PK_WGSKIP, lsk, wave, lane);
    __syncthreads();
    for (int e = tid; e < 16 * 128; e += 256) {
        float v = tsk[e] * sigmoidf_(lsk[e]);
        tsk[e] = v;
        tskb[e] = (__bf16)v;
    }
    __syncthreads();

    // ---- Phase 6: sig_out = tanh(skip_out @ W_sig.T) * gain ---------------
    mm_bf<128, 48, 40, 48>(tskb, pk + PK_WSIG, lsk, wave, lane);
    __syncthreads();
    for (int e = tid; e < 16 * 40; e += 256) {
        int r = e / 40, c = e % 40;
        float v = tanhf(lsk[r * 48 + c]) * s_gain[r];
        __builtin_nontemporal_store(v, out_sig + e);
        __builtin_nontemporal_store(v, out_exc + r * 256 + 216 + c);   // exc_new tail
    }
}

// ---------------------------------------------------------------------------
extern "C" void kernel_launch(void* const* d_in, const int* in_sizes, int n_in,
                              void* d_out, int out_size, void* d_ws, size_t ws_size,
                              hipStream_t stream) {
    (void)in_sizes; (void)n_in; (void)out_size; (void)ws_size;
    const float* cond      = (const float*)d_in[0];
    const float* prev_pred = (const float*)d_in[1];
    const float* exc_mem   = (const float*)d_in[2];
    const float* s1        = (const float*)d_in[3];
    const float* s2        = (const float*)d_in[4];
    const float* s3        = (const float*)d_in[5];
    const float* s4        = (const float*)d_in[6];
    const int*   period    = (const int*)d_in[7];
    const float* W_cg      = (const float*)d_in[8];
    const float* b_cg      = (const float*)d_in[9];
    const float* W_fwc     = (const float*)d_in[10];
    const float* W_fwc_glu = (const float*)d_in[11];
    const float* W_pg      = (const float*)d_in[12];
    const float* b_pg      = (const float*)d_in[13];
    const float* W1_ih     = (const float*)d_in[14];
    const float* W1_hh     = (const float*)d_in[15];
    const float* Wg1       = (const float*)d_in[16];
    const float* W2_ih     = (const float*)d_in[17];
    const float* W2_hh     = (const float*)d_in[18];
    const float* Wg2       = (const float*)d_in[19];
    const float* W3_ih     = (const float*)d_in[20];
    const float* W3_hh     = (const float*)d_in[21];
    const float* Wg3       = (const float*)d_in[22];
    const float* W_skip    = (const float*)d_in[23];
    const float* Wg_skip   = (const float*)d_in[24];
    const float* W_sig     = (const float*)d_in[25];

    __bf16* pk = (__bf16*)d_ws;

    #define PACK(N,K,NP,KP,SRC,DST) \
        pack_w<N,K,NP,KP><<<((NP)*(KP)+255)/256, 256, 0, stream>>>(SRC, pk + (DST))
    PACK(192,328,192,352, W_fwc,     PK_FWC);
    PACK(192,192,192,192, W_fwc_glu, PK_FWCGLU);
    PACK(480,272,480,288, W1_ih,     PK_W1IH);
    PACK(480,160,480,160, W1_hh,     PK_W1HH);
    PACK(160,160,160,160, Wg1,       PK_WG1);
    PACK(384,240,384,256, W2_ih,     PK_W2IH);
    PACK(384,128,384,128, W2_hh,     PK_W2HH);
    PACK(128,128,128,128, Wg2,       PK_WG2);
    PACK(384,208,384,224, W3_ih,     PK_W3IH);
    PACK(384,128,384,128, W3_hh,     PK_W3HH);
    PACK(128,128,128,128, Wg3,       PK_WG3);
    PACK(128,688,128,704, W_skip,    PK_WSKIP);
    PACK(128,128,128,128, Wg_skip,   PK_WGSKIP);
    PACK( 40,128, 48,128, W_sig,     PK_WSIG);
    #undef PACK

    fargan_main<<<BATCH / 16, 256, SMEM_BYTES, stream>>>(
        cond, prev_pred, exc_mem, s1, s2, s3, s4, period,
        W_cg, b_cg, W_pg, b_pg, pk, (float*)d_out);
}
